// SwinBlock_33930241638728
// MI455X (gfx1250) — matrix operand order repaired and checked by reading, not compiled, not run on hardware
//
#include <hip/hip_runtime.h>

// ---------------------------------------------------------------------------
// Swin block x2 for MI455X (gfx1250, wave32, WMMA 16x16x32 bf16)
// B=2, C=192, H=W=256, window 8x8 (64 tokens), heads=6, head_dim=32
// ---------------------------------------------------------------------------

typedef __bf16 bf16x16 __attribute__((ext_vector_type(16)));
typedef __bf16 bf16x8  __attribute__((ext_vector_type(8)));
typedef float  f32x8   __attribute__((ext_vector_type(8)));

#define CC 192          // channels
#define HID 768         // mlp hidden
#define NTOK 131072     // 2*256*256 tokens

// float -> bf16 round-to-nearest-even
static __device__ inline unsigned short f2bf(float f) {
    unsigned int u = __float_as_uint(f);
    u += 0x7FFFu + ((u >> 16) & 1u);
    return (unsigned short)(u >> 16);
}

static __device__ inline f32x8 fz() {
    f32x8 z = {0.f, 0.f, 0.f, 0.f, 0.f, 0.f, 0.f, 0.f};
    return z;
}

static __device__ inline f32x8 wmma_bf16(bf16x16 a, bf16x16 b, f32x8 c) {
    // emits v_wmma_f32_16x16x32_bf16
    return __builtin_amdgcn_wmma_f32_16x16x32_bf16(false, a, false, b, (short)0, c,
                                                   false, false);
}

// A-matrix fragment (16x32 bf16, row-major source, stride in halves).
// Lane l: row = row0 + (l&15); elements 0..15 = K k0+(l<16?0:16)+0..15
// (contiguous k order, matching B fragments).
static __device__ inline bf16x16 load_a(const unsigned short* base, int stride,
                                        int row0, int k0, int lane) {
    const unsigned short* p =
        base + (row0 + (lane & 15)) * stride + k0 + ((lane >> 4) << 4);
    union { bf16x16 v; bf16x8 h[2]; } u;
    u.h[0] = *(const bf16x8*)p;
    u.h[1] = *(const bf16x8*)(p + 8);
    return u.v;
}

// B fragment from 16 contiguous halves (lane-specific pointer precomputed).
static __device__ inline bf16x16 load_b16(const unsigned short* p) {
    union { bf16x16 v; bf16x8 h[2]; } u;
    u.h[0] = *(const bf16x8*)p;
    u.h[1] = *(const bf16x8*)(p + 8);
    return u.v;
}

// B fragment from packed weights: layout [nt][ks][lane][16]
static __device__ inline bf16x16 load_bpk(const unsigned short* wp, int nt, int ks,
                                          int KS, int lane) {
    return load_b16(wp + ((((nt * KS) + ks) * 32 + lane) << 4));
}

// Prefetch a weight image into cache (global_prefetch_b8; no LOADcnt cost).
static __device__ inline void prefetch_w(const unsigned short* p, int bytes, int tid) {
    for (int o = tid * 256; o < bytes; o += 256 * 256)
        __builtin_prefetch((const char*)p + o, 0, 1);
}

// ---------------------------------------------------------------------------
// Weight packing: W (KxN f32 row-major) -> bf16 fragments [N/16][K/32][32][16]
// element [nt][ks][lane][i] = W[ks*32 + (lane<16?0:16) + i][nt*16 + (lane&15)]
// ---------------------------------------------------------------------------
__global__ void pack_kernel(const float* __restrict__ w, unsigned short* __restrict__ o,
                            int K, int N) {
    int idx = blockIdx.x * 256 + threadIdx.x;
    if (idx >= K * N) return;
    int i    = idx & 15;
    int lane = (idx >> 4) & 31;
    int grp  = idx >> 9;            // nt*KS + ks
    int KS   = K >> 5;
    int nt   = grp / KS;
    int ks   = grp - nt * KS;
    int n = nt * 16 + (lane & 15);
    int k = ks * 32 + ((lane >> 4) << 4) + i;
    o[idx] = f2bf(w[k * N + n]);
}

// ---------------------------------------------------------------------------
// Window attention: x + proj(softmax(qk^T/sqrt(d) + bias [+mask]) v)
// One workgroup = one 8x8 window (2048 workgroups). 8 waves.
// ---------------------------------------------------------------------------
template <int SHIFTED, int NCHW_IN>
__global__ void __launch_bounds__(256)
attn_kernel(const float* xin, float* xout,
            const unsigned short* __restrict__ qkvw, const float* __restrict__ qkvb,
            const float* __restrict__ rpb,
            const unsigned short* __restrict__ projw, const float* __restrict__ projb,
            const float* __restrict__ lng, const float* __restrict__ lnb) {
    __shared__ alignas(16) float          s_xres[64 * CC];    // 48 KB residual
    __shared__ alignas(16) unsigned short s_xln[64 * CC];     // 24 KB ln-out / attn-out
    __shared__ alignas(16) unsigned short s_qkv[64 * 576];    // 72 KB q|k|v bf16
    __shared__ alignas(16) unsigned short s_probs[8 * 16 * 64]; // 16 KB wave-private

    const int tid = threadIdx.x;
    const int lane = tid & 31;
    const int wave = tid >> 5;
    const int wg = blockIdx.x;
    const int bb = wg >> 10;
    const int win = wg & 1023;
    const int wy = win >> 5, wx = win & 31;
    const int sh = SHIFTED ? 4 : 0;

    // warm the weight stream while we gather the window
    prefetch_w(qkvw, 192 * 576 * 2, tid);
    prefetch_w(projw, 192 * 192 * 2, tid);

    // ---- gather window (fused transpose + cyclic shift), float4 vectorized ----
    if (NCHW_IN) {
        // only instantiated with SHIFTED=0: gx base = wx*8 (16B-aligned rows of 8)
        for (int idx = tid; idx < CC * 8 * 2; idx += 256) {
            int c = idx >> 4;
            int rem = idx & 15;
            int row = rem >> 1, half = rem & 1;
            int gy = (wy * 8 + row + sh) & 255;
            float4 v = *(const float4*)(xin + (((bb * CC + c) * 256 + gy) * 256) +
                                        wx * 8 + half * 4);
            int t = row * 8 + half * 4;
            s_xres[(t + 0) * CC + c] = v.x;
            s_xres[(t + 1) * CC + c] = v.y;
            s_xres[(t + 2) * CC + c] = v.z;
            s_xres[(t + 3) * CC + c] = v.w;
        }
    } else {
        for (int idx = tid; idx < 64 * 48; idx += 256) {
            int t = idx / 48;
            int q = idx - t * 48;
            int gy = (wy * 8 + (t >> 3) + sh) & 255;
            int gx = (wx * 8 + (t & 7) + sh) & 255;
            float4 v = ((const float4*)(xin + ((bb * 256 + gy) * 256 + gx) * CC))[q];
            *(float4*)(s_xres + t * CC + q * 4) = v;
        }
    }
    __syncthreads();

    // ---- LayerNorm 1 (4 lanes per token) ----
    {
        int t = tid >> 2, sub = tid & 3;
        float s = 0.f, s2 = 0.f;
        for (int c = sub; c < CC; c += 4) {
            float v = s_xres[t * CC + c];
            s += v; s2 += v * v;
        }
        s  += __shfl_xor(s, 1, 4);  s  += __shfl_xor(s, 2, 4);
        s2 += __shfl_xor(s2, 1, 4); s2 += __shfl_xor(s2, 2, 4);
        float mu = s * (1.f / CC);
        float rstd = rsqrtf(s2 * (1.f / CC) - mu * mu + 1e-5f);
        for (int c = sub; c < CC; c += 4)
            s_xln[t * CC + c] = f2bf((s_xres[t * CC + c] - mu) * rstd * lng[c] + lnb[c]);
    }
    __syncthreads();

    // ---- QKV GEMM: (64x192) @ (192x576), tiles 4 x 36 ----
    for (int tile = wave; tile < 144; tile += 8) {
        int mt = tile / 36, nt = tile % 36;
        f32x8 acc = fz();
#pragma unroll
        for (int ks = 0; ks < 6; ks++) {
            bf16x16 a = load_a(s_xln, CC, mt * 16, ks * 32, lane);
            bf16x16 b = load_bpk(qkvw, nt, ks, 6, lane);
            acc = wmma_bf16(a, b, acc);
        }
        int cn = lane & 15, hl = lane >> 4;
        int col = nt * 16 + cn;
        float bias = qkvb[col];
#pragma unroll
        for (int r = 0; r < 8; r++)
            s_qkv[(mt * 16 + r + 8 * hl) * 576 + col] = f2bf(acc[r] + bias);
    }
    __syncthreads();

    // ---- attention per (head, mtile): 6 heads x 4 mtiles = 24 units ----
    const float scale = 0.17677669529663687f;  // 32^-0.5
    for (int unit = wave; unit < 24; unit += 8) {
        int head = unit >> 2, mt = unit & 3;
        int cn = lane & 15, hl = lane >> 4;

        // sim = q k^T : 16x64 block, hd=32 = one WMMA K-step
        bf16x16 aq = load_a(s_qkv, 576, mt * 16, head * 32, lane);
        f32x8 c[4];
#pragma unroll
        for (int nt = 0; nt < 4; nt++) {
            const unsigned short* kp =
                s_qkv + (nt * 16 + cn) * 576 + 192 + head * 32 + ((lane < 16) ? 0 : 16);
            c[nt] = wmma_bf16(aq, load_b16(kp), fz());
        }
        // bias + mask + softmax over rows (row spans 16 lanes of a half-wave)
#pragma unroll
        for (int r = 0; r < 8; r++) {
            int q = mt * 16 + r + 8 * hl;
            float mx = -3.0e38f;
#pragma unroll
            for (int nt = 0; nt < 4; nt++) {
                int kk = nt * 16 + cn;
                int rel = (((q >> 3) - (kk >> 3) + 7) * 15) + ((q & 7) - (kk & 7) + 7);
                float v = c[nt][r] * scale + rpb[head * 225 + rel];
                if (SHIFTED) {
                    if ((wy == 31) && (((q ^ kk) >> 3) & 4)) v = -1e30f;
                    if ((wx == 31) && ((q ^ kk) & 4)) v = -1e30f;
                }
                c[nt][r] = v;
                mx = fmaxf(mx, v);
            }
            mx = fmaxf(mx, __shfl_xor(mx, 1, 16));
            mx = fmaxf(mx, __shfl_xor(mx, 2, 16));
            mx = fmaxf(mx, __shfl_xor(mx, 4, 16));
            mx = fmaxf(mx, __shfl_xor(mx, 8, 16));
            float sm = 0.f;
#pragma unroll
            for (int nt = 0; nt < 4; nt++) {
                float e = __expf(c[nt][r] - mx);
                c[nt][r] = e;
                sm += e;
            }
            sm += __shfl_xor(sm, 1, 16); sm += __shfl_xor(sm, 2, 16);
            sm += __shfl_xor(sm, 4, 16); sm += __shfl_xor(sm, 8, 16);
            float inv = 1.f / sm;
#pragma unroll
            for (int nt = 0; nt < 4; nt++) c[nt][r] *= inv;
        }
        // probs -> bf16 A-layout via wave-private LDS transpose
        unsigned short* pw = s_probs + wave * 1024;
#pragma unroll
        for (int nt = 0; nt < 4; nt++)
#pragma unroll
            for (int r = 0; r < 8; r++)
                pw[(r + 8 * hl) * 64 + nt * 16 + cn] = f2bf(c[nt][r]);

        // out = probs @ v : 16x32, K=64 keys = 2 WMMA K-steps
#pragma unroll
        for (int n2 = 0; n2 < 2; n2++) {
            f32x8 o = fz();
#pragma unroll
            for (int ks = 0; ks < 2; ks++) {
                bf16x16 a = load_a(pw, 64, 0, ks * 32, lane);
                union { bf16x16 v; unsigned short u[16]; } bv;
                int tok0 = ks * 32 + ((lane < 16) ? 0 : 16);
#pragma unroll
                for (int i = 0; i < 16; i++)
                    bv.u[i] = s_qkv[(tok0 + i) * 576 + 384 + head * 32 + n2 * 16 + cn];
                o = wmma_bf16(a, bv.v, o);
            }
#pragma unroll
            for (int r = 0; r < 8; r++)
                s_xln[(mt * 16 + r + 8 * hl) * CC + head * 32 + n2 * 16 + cn] =
                    f2bf(o[r]);
        }
    }
    __syncthreads();

    // ---- proj GEMM (64x192)@(192x192) + bias + residual, store NHWC ----
    for (int tile = wave; tile < 48; tile += 8) {
        int mt = tile / 12, nt = tile % 12;
        f32x8 acc = fz();
#pragma unroll
        for (int ks = 0; ks < 6; ks++) {
            bf16x16 a = load_a(s_xln, CC, mt * 16, ks * 32, lane);
            bf16x16 b = load_bpk(projw, nt, ks, 6, lane);
            acc = wmma_bf16(a, b, acc);
        }
        int cn = lane & 15, hl = lane >> 4;
        int col = nt * 16 + cn;
        float bias = projb[col];
#pragma unroll
        for (int r = 0; r < 8; r++) {
            int t = mt * 16 + r + 8 * hl;
            int gy = (wy * 8 + (t >> 3) + sh) & 255;
            int gx = (wx * 8 + (t & 7) + sh) & 255;
            xout[((bb * 256 + gy) * 256 + gx) * CC + col] =
                s_xres[t * CC + col] + bias + acc[r];
        }
    }
}

// ---------------------------------------------------------------------------
// MLP: x + gelu(ln2(x) @ w1 + b1) @ w2 + b2    (64 tokens per workgroup)
// ---------------------------------------------------------------------------
template <int NCHW_OUT>
__global__ void __launch_bounds__(256)
mlp_kernel(const float* xin, float* xout,
           const float* __restrict__ lng, const float* __restrict__ lnb,
           const unsigned short* __restrict__ w1p, const float* __restrict__ b1,
           const unsigned short* __restrict__ w2p, const float* __restrict__ b2) {
    __shared__ alignas(16) unsigned short s_xln[64 * CC];   // 24 KB
    __shared__ alignas(16) unsigned short s_h[64 * HID];    // 96 KB

    const int tid = threadIdx.x;
    const int lane = tid & 31;
    const int wave = tid >> 5;
    const int t0 = blockIdx.x * 64;

    prefetch_w(w1p, 192 * 768 * 2, tid);
    prefetch_w(w2p, 768 * 192 * 2, tid);

    // ---- LayerNorm 2 ----
    {
        int t = tid >> 2, sub = tid & 3;
        const float* xp = xin + (t0 + t) * CC;
        float s = 0.f, s2 = 0.f;
        for (int c = sub; c < CC; c += 4) {
            float v = xp[c];
            s += v; s2 += v * v;
        }
        s  += __shfl_xor(s, 1, 4);  s  += __shfl_xor(s, 2, 4);
        s2 += __shfl_xor(s2, 1, 4); s2 += __shfl_xor(s2, 2, 4);
        float mu = s * (1.f / CC);
        float rstd = rsqrtf(s2 * (1.f / CC) - mu * mu + 1e-5f);
        for (int c = sub; c < CC; c += 4)
            s_xln[t * CC + c] = f2bf((xp[c] - mu) * rstd * lng[c] + lnb[c]);
    }
    __syncthreads();

    // ---- GEMM1 (64x192)@(192x768) + GELU(exact erf) ----
    for (int tile = wave; tile < 4 * 48; tile += 8) {
        int mt = tile / 48, nt = tile % 48;
        f32x8 acc = fz();
#pragma unroll
        for (int ks = 0; ks < 6; ks++) {
            bf16x16 a = load_a(s_xln, CC, mt * 16, ks * 32, lane);
            bf16x16 b = load_bpk(w1p, nt, ks, 6, lane);
            acc = wmma_bf16(a, b, acc);
        }
        int cn = lane & 15, hl = lane >> 4;
        int col = nt * 16 + cn;
        float bias = b1[col];
#pragma unroll
        for (int r = 0; r < 8; r++) {
            float x = acc[r] + bias;
            float g = 0.5f * x * (1.f + erff(x * 0.70710678118654752f));
            s_h[(mt * 16 + r + 8 * hl) * HID + col] = f2bf(g);
        }
    }
    __syncthreads();

    // ---- GEMM2 (64x768)@(768x192) + bias + residual ----
    for (int tile = wave; tile < 4 * 12; tile += 8) {
        int mt = tile / 12, nt = tile % 12;
        f32x8 acc = fz();
#pragma unroll
        for (int ks = 0; ks < 24; ks++) {
            bf16x16 a = load_a(s_h, HID, mt * 16, ks * 32, lane);
            bf16x16 b = load_bpk(w2p, nt, ks, 24, lane);
            acc = wmma_bf16(a, b, acc);
        }
        int cn = lane & 15, hl = lane >> 4;
        int col = nt * 16 + cn;
        float bias = b2[col];
#pragma unroll
        for (int r = 0; r < 8; r++) {
            int tg = t0 + mt * 16 + r + 8 * hl;
            float v = xin[tg * CC + col] + bias + acc[r];
            if (NCHW_OUT) {
                int bb = tg >> 16;
                int rem = tg & 65535;
                int gy = rem >> 8, gx = rem & 255;
                xout[((bb * CC + col) * 256 + gy) * 256 + gx] = v;
            } else {
                xout[tg * CC + col] = v;  // in-place NHWC (disjoint per thread)
            }
        }
    }
}

// ---------------------------------------------------------------------------
extern "C" void kernel_launch(void* const* d_in, const int* in_sizes, int n_in,
                              void* d_out, int out_size, void* d_ws, size_t ws_size,
                              hipStream_t stream) {
    (void)in_sizes; (void)n_in; (void)out_size; (void)ws_size;
    const float* x = (const float*)d_in[0];
    const float* p[27];
    for (int i = 0; i < 27; i++) p[i] = (const float*)d_in[i];

    char* ws = (char*)d_ws;
    float* bufA = (float*)ws;                         // NHWC activations, 100.66 MB
    size_t off = (size_t)NTOK * CC * 4;
    auto nxt = [&](size_t bytes) {
        char* r = ws + off;
        off += (bytes + 255) & ~(size_t)255;
        return (unsigned short*)r;
    };
    unsigned short* qkvp1 = nxt((size_t)192 * 576 * 2);
    unsigned short* prjp1 = nxt((size_t)192 * 192 * 2);
    unsigned short* w1p1  = nxt((size_t)192 * 768 * 2);
    unsigned short* w2p1  = nxt((size_t)768 * 192 * 2);
    unsigned short* qkvp2 = nxt((size_t)192 * 576 * 2);
    unsigned short* prjp2 = nxt((size_t)192 * 192 * 2);
    unsigned short* w1p2  = nxt((size_t)192 * 768 * 2);
    unsigned short* w2p2  = nxt((size_t)768 * 192 * 2);

    auto pack = [&](const float* w, unsigned short* o, int K, int N) {
        int total = K * N;
        pack_kernel<<<(total + 255) / 256, 256, 0, stream>>>(w, o, K, N);
    };
    pack(p[3],  qkvp1, 192, 576);
    pack(p[6],  prjp1, 192, 192);
    pack(p[10], w1p1,  192, 768);
    pack(p[12], w2p1,  768, 192);
    pack(p[16], qkvp2, 192, 576);
    pack(p[19], prjp2, 192, 192);
    pack(p[23], w1p2,  192, 768);
    pack(p[25], w2p2,  768, 192);

    // Block 1 (not shifted): NCHW input -> NHWC bufA, then MLP in place
    attn_kernel<0, 1><<<2048, 256, 0, stream>>>(x, bufA, qkvp1, p[4], p[5], prjp1,
                                                p[7], p[1], p[2]);
    mlp_kernel<0><<<2048, 256, 0, stream>>>(bufA, bufA, p[8], p[9], w1p1, p[11],
                                            w2p1, p[13]);
    // Block 2 (shifted): in-place on bufA (each WG owns its rolled window)
    attn_kernel<1, 0><<<2048, 256, 0, stream>>>(bufA, bufA, qkvp2, p[17], p[18],
                                                prjp2, p[20], p[14], p[15]);
    // Final MLP writes NCHW straight to d_out
    mlp_kernel<1><<<2048, 256, 0, stream>>>(bufA, (float*)d_out, p[21], p[22], w1p2,
                                            p[24], w2p2, p[26]);
}
